// Model_6416681140655
// MI455X (gfx1250) — matrix operand order repaired and checked by reading, compiled, not run on hardware
//
#include <hip/hip_runtime.h>
#include <hip/hip_bf16.h>

#define HIDDEN 64
#define LEAKY  0.01f

typedef _Float16 v16h __attribute__((ext_vector_type(16)));
typedef _Float16 v8h  __attribute__((ext_vector_type(8)));
typedef _Float16 v4h  __attribute__((ext_vector_type(4)));
typedef float    v8f  __attribute__((ext_vector_type(8)));

// ---- monotonic float <-> uint mapping so we can use integer atomic max ----
__device__ __forceinline__ unsigned f32_to_ordered(float f) {
    unsigned u = __float_as_uint(f);
    return (u & 0x80000000u) ? ~u : (u | 0x80000000u);
}
__device__ __forceinline__ float ordered_to_f32(unsigned u) {
    unsigned b = (u & 0x80000000u) ? (u ^ 0x80000000u) : ~u;
    return __uint_as_float(b);
}

// ---------------------------------------------------------------------------
// Init: zero accumulators, set segment-max identity, convert attn_w -> f16
// ---------------------------------------------------------------------------
__global__ void k_init(float* __restrict__ acc, float* __restrict__ denom,
                       unsigned* __restrict__ m_u,
                       const float* __restrict__ attn_w,
                       _Float16* __restrict__ wh, int N) {
    int idx = blockIdx.x * blockDim.x + threadIdx.x;
    if (idx < N * HIDDEN) acc[idx] = 0.0f;
    if (idx < N) { denom[idx] = 0.0f; m_u[idx] = 0u; }   // 0 == ordered(< -inf)
    if (idx < 2 * HIDDEN) wh[idx] = (_Float16)attn_w[idx];
}

// ---------------------------------------------------------------------------
// Gather h = emb[in_feat]; write f32 copy (for accumulate/finalize) and
// f16 copy (WMMA B operand). 16 threads per node, float4 per thread.
// ---------------------------------------------------------------------------
__global__ void k_gather(const int* __restrict__ in_feat,
                         const float* __restrict__ emb,
                         float* __restrict__ h, _Float16* __restrict__ hh,
                         int N) {
    int idx = blockIdx.x * blockDim.x + threadIdx.x;
    if (idx >= N * (HIDDEN / 4)) return;
    int n = idx >> 4, g = (idx & 15) * 4;
    int f = in_feat[n];
    float4 v = *(const float4*)(emb + (size_t)f * HIDDEN + g);
    *(float4*)(h + (size_t)n * HIDDEN + g) = v;
    v4h hv = { (_Float16)v.x, (_Float16)v.y, (_Float16)v.z, (_Float16)v.w };
    *(v4h*)(hh + (size_t)n * HIDDEN + g) = hv;
}

// ---------------------------------------------------------------------------
// Edge logits via WMMA: one wave handles 16 edges.
//   A[m,k] = w[chunk*32 + k]   (replicated over rows m)
//   B[k,n] = z_{edge n}[chunk*32 + k]
//   => D[m,n] = logit(edge n) for every m; D vgpr0 holds e(edge lane&15).
// K chunks: [w0..31 x zsrc0..31][w32..63 x zsrc32..63][w64..95 x zdst0..31]
//           [w96..127 x zdst32..63]  -> 4 x v_wmma_f32_16x16x32_f16
// ---------------------------------------------------------------------------
__global__ void __launch_bounds__(256) k_edge_logits(
    const int* __restrict__ src, const int* __restrict__ dst,
    const _Float16* __restrict__ hh, const _Float16* __restrict__ wh,
    float* __restrict__ e_arr, unsigned* __restrict__ m_u, int E) {
    const int lane = threadIdx.x & 31;
    const int wave = threadIdx.x >> 5;
    const int base = (blockIdx.x * 8 + wave) * 16;
    const int er   = base + (lane & 15);
    const int edge = (er < E) ? er : (E - 1);          // clamp; stores masked
    const int s = src[edge];
    const int d = dst[edge];
    // A loaded via kb, contiguous k: lanes<16 hold K = 0..15,
    //                        lanes>=16 hold K = 16..31 (ka below is unused)
    const int ka = (lane < 16) ? 0 : 8;
    // B uses the same contiguous k: lanes<16 hold K = 0..15, lanes>=16 K = 16..31
    const int kb = (lane < 16) ? 0 : 16;

    v8f acc = {};
#pragma unroll
    for (int chunk = 0; chunk < 4; ++chunk) {
        const int wbase = chunk * 32;           // offset into 128-wide w
        const int row   = (chunk < 2) ? s : d;  // src half then dst half
        const int zbase = (chunk & 1) * 32;     // feature base within row
        v8h wlo = *(const v8h*)(wh + wbase + kb);
        v8h whi = *(const v8h*)(wh + wbase + kb + 8);
        v16h a;
#pragma unroll
        for (int j = 0; j < 8; ++j) { a[j] = wlo[j]; a[8 + j] = whi[j]; }
        v16h b = *(const v16h*)(hh + (size_t)row * HIDDEN + zbase + kb);
        acc = __builtin_amdgcn_wmma_f32_16x16x32_f16(
            false, a, false, b, (short)0, acc, false, false);
    }
    // D[M=0 or 8, N=lane&15] lives in acc[0] of every lane.
    float e = acc[0];
    e = (e > 0.0f) ? e : LEAKY * e;                    // leaky_relu
    if (lane < 16 && er < E) {
        e_arr[er] = e;
        atomicMax(m_u + d, f32_to_ordered(e));         // segment max
    }
}

// ---------------------------------------------------------------------------
// Softmax weight + scatter accumulate: one wave per edge, lane l owns
// features 2l, 2l+1. acc[dst,:] += p * z_src ; denom[dst] += p.
// ---------------------------------------------------------------------------
__global__ void __launch_bounds__(256) k_edge_scatter(
    const int* __restrict__ src, const int* __restrict__ dst,
    const float* __restrict__ h, const float* __restrict__ e_arr,
    const unsigned* __restrict__ m_u,
    float* __restrict__ denom, float* __restrict__ acc, int E) {
    const int lane = threadIdx.x & 31;
    const int edge = blockIdx.x * 8 + (threadIdx.x >> 5);
    if (edge >= E) return;
    const int s = src[edge];
    const int d = dst[edge];
    const float ev = e_arr[edge];
    const float mv = ordered_to_f32(m_u[d]);
    const float p  = expf(ev - mv);                    // <= 1, stable
    if (lane == 0) atomicAdd(denom + d, p);
    const float2 z = *(const float2*)(h + (size_t)s * HIDDEN + lane * 2);
    float* ap = acc + (size_t)d * HIDDEN + lane * 2;
    atomicAdd(ap,     p * z.x);
    atomicAdd(ap + 1, p * z.y);
}

// ---------------------------------------------------------------------------
// Finalize: agg = acc/denom, empty-mailbox keeps h, then ELU.
// ---------------------------------------------------------------------------
__global__ void k_finalize(const float* __restrict__ h,
                           const float* __restrict__ acc,
                           const float* __restrict__ denom,
                           float* __restrict__ out, int N) {
    int idx = blockIdx.x * blockDim.x + threadIdx.x;
    if (idx >= N * HIDDEN) return;
    int n = idx >> 6;
    float den = denom[n];
    float v = (den > 0.0f) ? acc[idx] / fmaxf(den, 1e-20f) : h[idx];
    out[idx] = (v > 0.0f) ? v : expm1f(v);             // elu
}

extern "C" void kernel_launch(void* const* d_in, const int* in_sizes, int n_in,
                              void* d_out, int out_size, void* d_ws, size_t ws_size,
                              hipStream_t stream) {
    const int*   in_feat = (const int*)d_in[0];
    const int*   src     = (const int*)d_in[1];
    const int*   dst     = (const int*)d_in[2];
    const float* emb     = (const float*)d_in[3];
    const float* attn_w  = (const float*)d_in[4];
    const int N = in_sizes[0];
    const int E = in_sizes[1];
    (void)n_in; (void)out_size; (void)ws_size;

    // workspace carve (256 B aligned): ~71 MB total
    char* p = (char*)d_ws;
    auto carve = [&](size_t bytes) -> char* {
        char* r = p; p += (bytes + 255) & ~(size_t)255; return r;
    };
    float*    h     = (float*)   carve((size_t)N * HIDDEN * sizeof(float));
    _Float16* hh    = (_Float16*)carve((size_t)N * HIDDEN * sizeof(_Float16));
    float*    e_arr = (float*)   carve((size_t)E * sizeof(float));
    unsigned* m_u   = (unsigned*)carve((size_t)N * sizeof(unsigned));
    float*    denom = (float*)   carve((size_t)N * sizeof(float));
    float*    acc   = (float*)   carve((size_t)N * HIDDEN * sizeof(float));
    _Float16* wh    = (_Float16*)carve(2 * HIDDEN * sizeof(_Float16));

    const dim3 blk(256);
    k_init<<<dim3((unsigned)((N * HIDDEN + 255) / 256)), blk, 0, stream>>>(
        acc, denom, m_u, attn_w, wh, N);
    k_gather<<<dim3((unsigned)((N * (HIDDEN / 4) + 255) / 256)), blk, 0, stream>>>(
        in_feat, emb, h, hh, N);
    k_edge_logits<<<dim3((unsigned)((E + 127) / 128)), blk, 0, stream>>>(
        src, dst, hh, wh, e_arr, m_u, E);
    k_edge_scatter<<<dim3((unsigned)((E + 7) / 8)), blk, 0, stream>>>(
        src, dst, h, e_arr, m_u, denom, acc, E);
    k_finalize<<<dim3((unsigned)((N * HIDDEN + 255) / 256)), blk, 0, stream>>>(
        h, acc, denom, (float*)d_out, N);
}
